// GATModel_26491358282159
// MI455X (gfx1250) — matrix operand order repaired and checked by reading, not compiled, not run on hardware
//
#include <hip/hip_runtime.h>
#include <hip/hip_bf16.h>

// ---------------- problem constants (match reference) ----------------
#define N_  100000
#define E_  1600000
#define ET_ (E_ + N_)      // edges + self loops
#define FIN 256
#define H_  14
#define C_  7
#define HC  98             // H_*C_
#define NTILES_M 6250      // N_/16
#define NTILES_N 7         // 112/16 (98 padded to 112)
#define WPAD 112           // padded output columns

typedef __attribute__((ext_vector_type(16))) __bf16          v16bf;
typedef __attribute__((ext_vector_type(8)))  float           v8f;
typedef __attribute__((ext_vector_type(4)))  float           v4f;
typedef __attribute__((ext_vector_type(4)))  unsigned int    v4u;

__device__ __forceinline__ void atomicMaxF(float* a, float v) {
    // ordering-safe float max via integer atomics; requires init = -inf
    if (v >= 0.0f) atomicMax((int*)a, __float_as_int(v));
    else           atomicMin((unsigned int*)a, __float_as_uint(v));
}

__device__ __forceinline__ void edge_sd(int e, const int* ei, int& s, int& d) {
    if (e < E_) { s = ei[e]; d = ei[E_ + e]; }
    else        { s = e - E_; d = s; }            // self loop
}

#define NEG_INF __int_as_float(0xFF800000)

// ---------------- W1 -> bf16, transposed + padded [112][256] ----------------
__global__ void k_prep_w1(const float* __restrict__ W1, __bf16* __restrict__ Wt) {
    int idx = blockIdx.x * blockDim.x + threadIdx.x;
    if (idx >= WPAD * FIN) return;
    int j = idx >> 8;      // output column of layer-1 (0..111)
    int k = idx & 255;     // reduction index
    float v = (j < HC) ? W1[k * HC + j] : 0.0f;
    Wt[idx] = (__bf16)v;
}

// ---------------- Layer-1 GEMM: xh = x @ W1, bf16 WMMA, f32 accumulate ------
// B (112x256 bf16 = 57KB) is staged once per workgroup into LDS (320KB/WGP on
// CDNA5), so the WMMA inner loop feeds from ds_load_b128 instead of stalling on
// global loads. One wave owns a 16-row strip and all 7 column tiles: the A
// fragment is converted once per k-step (v_cvt_pk_bf16_f32) and reused across
// 7 WMMAs; x is read exactly once from HBM.
__global__ __launch_bounds__(256)
void k_gemm1(const float* __restrict__ x, const __bf16* __restrict__ Wt,
             float* __restrict__ xh) {
    __shared__ __bf16 Bs[WPAD * FIN];     // 57344 bytes

    // cooperative 128-bit copy of W1(bf16,transposed) into LDS
    {
        const v4u* srcv = (const v4u*)Wt;
        v4u*       dstv = (v4u*)Bs;
        int tid = threadIdx.x;
#pragma unroll
        for (int i = 0; i < (WPAD * FIN * 2 / 16) / 256; ++i)
            dstv[tid + i * 256] = srcv[tid + i * 256];
    }
    __syncthreads();

    int wave = (blockIdx.x * blockDim.x + threadIdx.x) >> 5;
    int lane = threadIdx.x & 31;
    if (wave >= NTILES_M) return;
    int half = lane >> 4;          // 0: lanes 0-15, 1: lanes 16-31
    int l    = lane & 15;
    int Mb = wave * 16;

    const float* arow = x + (size_t)(Mb + l) * FIN;

    v8f acc[NTILES_N];
#pragma unroll
    for (int t = 0; t < NTILES_N; ++t) acc[t] = (v8f){};

#pragma unroll
    for (int kk = 0; kk < FIN; kk += 32) {
        // ---- A fragment (16x32 bf16): lane<16 holds K=kk+0..15,
        //      lane>=16 holds K=kk+16..31 (row M = Mb+l) ----
        int p = kk + half * 16;
        v4f a0 = *(const v4f*)(arow + p);
        v4f a1 = *(const v4f*)(arow + p + 4);
        v4f a2 = *(const v4f*)(arow + p + 8);
        v4f a3 = *(const v4f*)(arow + p + 12);
        v16bf A;
#pragma unroll
        for (int i = 0; i < 4; ++i) {
            A[i]      = (__bf16)a0[i];
            A[4 + i]  = (__bf16)a1[i];
            A[8 + i]  = (__bf16)a2[i];
            A[12 + i] = (__bf16)a3[i];
        }
        // ---- B fragments from LDS (32x16 bf16): column N = t*16+l,
        //      lane<16 holds K=kk+0..15, lane>=16 holds K=kk+16..31 ----
#pragma unroll
        for (int t = 0; t < NTILES_N; ++t) {
            const __bf16* bp = Bs + (t * 16 + l) * FIN + kk + half * 16;
            v16bf B = *(const v16bf*)bp;
            acc[t] = __builtin_amdgcn_wmma_f32_16x16x32_bf16(
                false, A, false, B, (short)0, acc[t], false, false);
        }
    }

#pragma unroll
    for (int t = 0; t < NTILES_N; ++t) {
        int col = t * 16 + l;
        if (col < HC) {
#pragma unroll
            for (int r = 0; r < 8; ++r)
                xh[(size_t)(Mb + half * 8 + r) * HC + col] = acc[t][r];
        }
    }
}

// ---------------- per-node attention logits: al = einsum(nhc,hc->nh) -------
__global__ void k_logits(const float* __restrict__ xh,
                         const float* __restrict__ asrc, const float* __restrict__ adst,
                         float* __restrict__ alS, float* __restrict__ alD) {
    int idx = blockIdx.x * blockDim.x + threadIdx.x;
    if (idx >= N_ * H_) return;
    int n = idx / H_, h = idx % H_;
    const float* v = xh + (size_t)n * HC + h * C_;
    float s = 0.0f, d = 0.0f;
#pragma unroll
    for (int c = 0; c < C_; ++c) {
        float t = v[c];
        s += t * asrc[h * C_ + c];
        d += t * adst[h * C_ + c];
    }
    alS[idx] = s;
    alD[idx] = d;
}

// ---------------- init segment buffers --------------------------------------
__global__ void k_init(float* __restrict__ amax, float* __restrict__ denom,
                       float* __restrict__ agg) {
    int idx = blockIdx.x * blockDim.x + threadIdx.x;
    if (idx >= N_ * HC) return;
    agg[idx] = NEG_INF;
    if (idx < N_ * H_) { amax[idx] = NEG_INF; denom[idx] = 0.0f; }
}

// ---------------- edge pass 1: segment max of leaky-relu logits -------------
__global__ void k_edge_max(const int* __restrict__ ei,
                           const float* __restrict__ alS, const float* __restrict__ alD,
                           float* __restrict__ amax) {
    int idx = blockIdx.x * blockDim.x + threadIdx.x;
    if (idx >= ET_ * H_) return;
    int e = idx / H_, h = idx % H_;
    int s, d; edge_sd(e, ei, s, d);
    float lg = alS[s * H_ + h] + alD[d * H_ + h];
    lg = (lg > 0.0f) ? lg : 0.2f * lg;
    atomicMaxF(&amax[d * H_ + h], lg);
}

// ---------------- edge pass 2: segment sum of exp ---------------------------
__global__ void k_edge_sum(const int* __restrict__ ei,
                           const float* __restrict__ alS, const float* __restrict__ alD,
                           const float* __restrict__ amax, float* __restrict__ denom) {
    int idx = blockIdx.x * blockDim.x + threadIdx.x;
    if (idx >= ET_ * H_) return;
    int e = idx / H_, h = idx % H_;
    int s, d; edge_sd(e, ei, s, d);
    float lg = alS[s * H_ + h] + alD[d * H_ + h];
    lg = (lg > 0.0f) ? lg : 0.2f * lg;
    atomicAdd(&denom[d * H_ + h], __expf(lg - amax[d * H_ + h]));
}

// ---------------- edge pass 3: weighted message + segment max ---------------
__global__ void k_edge_msg(const int* __restrict__ ei,
                           const float* __restrict__ alS, const float* __restrict__ alD,
                           const float* __restrict__ amax, const float* __restrict__ denom,
                           const float* __restrict__ xh, float* __restrict__ agg) {
    int idx = blockIdx.x * blockDim.x + threadIdx.x;
    if (idx >= ET_ * H_) return;
    int e = idx / H_, h = idx % H_;
    int s, d; edge_sd(e, ei, s, d);
    int dh = d * H_ + h;
    float lg = alS[s * H_ + h] + alD[dh];
    lg = (lg > 0.0f) ? lg : 0.2f * lg;
    float alpha = __expf(lg - amax[dh]) / denom[dh];
    const float* xs = xh + (size_t)s * HC + h * C_;
    float*       ad = agg + (size_t)d * HC + h * C_;
#pragma unroll
    for (int c = 0; c < C_; ++c)
        atomicMaxF(&ad[c], alpha * xs[c]);
}

// ---------------- layer-1 finish: mean over heads + bias + relu -------------
__global__ void k_finish1(const float* __restrict__ agg, const float* __restrict__ b,
                          float* __restrict__ h1) {
    int idx = blockIdx.x * blockDim.x + threadIdx.x;
    if (idx >= N_ * C_) return;
    int n = idx / C_, c = idx % C_;
    float s = 0.0f;
#pragma unroll
    for (int h = 0; h < H_; ++h) s += agg[(size_t)n * HC + h * C_ + c];
    float v = s * (1.0f / H_) + b[c];
    h1[idx] = v > 0.0f ? v : 0.0f;
}

// ---------------- layer-2 tiny GEMM (K=7, VALU is right) --------------------
__global__ void k_gemm2(const float* __restrict__ h1, const float* __restrict__ W2,
                        float* __restrict__ xh) {
    int idx = blockIdx.x * blockDim.x + threadIdx.x;
    if (idx >= N_ * HC) return;
    int n = idx / HC, j = idx % HC;
    float acc = 0.0f;
#pragma unroll
    for (int k = 0; k < C_; ++k) acc += h1[n * C_ + k] * W2[k * HC + j];
    xh[idx] = acc;
}

// ---------------- layer-2 finish: head mean + bias + log_softmax ------------
__global__ void k_finish2(const float* __restrict__ agg, const float* __restrict__ b,
                          float* __restrict__ out) {
    int n = blockIdx.x * blockDim.x + threadIdx.x;
    if (n >= N_) return;
    float v[C_];
    float m = NEG_INF;
#pragma unroll
    for (int c = 0; c < C_; ++c) {
        float s = 0.0f;
#pragma unroll
        for (int h = 0; h < H_; ++h) s += agg[(size_t)n * HC + h * C_ + c];
        v[c] = s * (1.0f / H_) + b[c];
        m = fmaxf(m, v[c]);
    }
    float se = 0.0f;
#pragma unroll
    for (int c = 0; c < C_; ++c) se += __expf(v[c] - m);
    float lse = __logf(se);
#pragma unroll
    for (int c = 0; c < C_; ++c) out[n * C_ + c] = v[c] - m - lse;
}

// ---------------- host-side launcher ----------------------------------------
extern "C" void kernel_launch(void* const* d_in, const int* in_sizes, int n_in,
                              void* d_out, int out_size, void* d_ws, size_t ws_size,
                              hipStream_t stream) {
    const float* x      = (const float*)d_in[0];
    const int*   ei     = (const int*)  d_in[1];
    const float* W1     = (const float*)d_in[2];
    const float* a_src1 = (const float*)d_in[3];
    const float* a_dst1 = (const float*)d_in[4];
    const float* b1     = (const float*)d_in[5];
    const float* W2     = (const float*)d_in[6];
    const float* a_src2 = (const float*)d_in[7];
    const float* a_dst2 = (const float*)d_in[8];
    const float* b2     = (const float*)d_in[9];
    float* out = (float*)d_out;

    // workspace bump allocator (256B aligned regions)
    char* base = (char*)d_ws;
    size_t off = 0;
    auto alloc = [&](size_t bytes) -> void* {
        off = (off + 255) & ~(size_t)255;
        void* p = base + off;
        off += bytes;
        return p;
    };
    float*  xh    = (float*)alloc((size_t)N_ * HC * 4);   // projections (reused L2)
    float*  alS   = (float*)alloc((size_t)N_ * H_ * 4);
    float*  alD   = (float*)alloc((size_t)N_ * H_ * 4);
    float*  amax  = (float*)alloc((size_t)N_ * H_ * 4);
    float*  denom = (float*)alloc((size_t)N_ * H_ * 4);
    float*  agg   = (float*)alloc((size_t)N_ * HC * 4);
    float*  h1    = (float*)alloc((size_t)N_ * C_ * 4);
    __bf16* Wt1   = (__bf16*)alloc((size_t)WPAD * FIN * 2);
    (void)ws_size; (void)in_sizes; (void)n_in; (void)out_size;

    const int TB = 256;
    auto blk = [&](long long n) { return (int)((n + TB - 1) / TB); };

    // ---- layer 1 ----
    k_prep_w1<<<blk(WPAD * FIN), TB, 0, stream>>>(W1, Wt1);
    k_gemm1<<<blk((long long)NTILES_M * 32), TB, 0, stream>>>(x, Wt1, xh);
    k_logits<<<blk((long long)N_ * H_), TB, 0, stream>>>(xh, a_src1, a_dst1, alS, alD);
    k_init<<<blk((long long)N_ * HC), TB, 0, stream>>>(amax, denom, agg);
    k_edge_max<<<blk((long long)ET_ * H_), TB, 0, stream>>>(ei, alS, alD, amax);
    k_edge_sum<<<blk((long long)ET_ * H_), TB, 0, stream>>>(ei, alS, alD, amax, denom);
    k_edge_msg<<<blk((long long)ET_ * H_), TB, 0, stream>>>(ei, alS, alD, amax, denom, xh, agg);
    k_finish1<<<blk((long long)N_ * C_), TB, 0, stream>>>(agg, b1, h1);

    // ---- layer 2 ----
    k_gemm2<<<blk((long long)N_ * HC), TB, 0, stream>>>(h1, W2, xh);
    k_logits<<<blk((long long)N_ * H_), TB, 0, stream>>>(xh, a_src2, a_dst2, alS, alD);
    k_init<<<blk((long long)N_ * HC), TB, 0, stream>>>(amax, denom, agg);
    k_edge_max<<<blk((long long)ET_ * H_), TB, 0, stream>>>(ei, alS, alD, amax);
    k_edge_sum<<<blk((long long)ET_ * H_), TB, 0, stream>>>(ei, alS, alD, amax, denom);
    k_edge_msg<<<blk((long long)ET_ * H_), TB, 0, stream>>>(ei, alS, alD, amax, denom, xh, agg);
    k_finish2<<<blk((long long)N_), TB, 0, stream>>>(agg, b2, out);
}
